// Set2Set_84018150244590
// MI455X (gfx1250) — matrix operand order repaired and checked by reading, not compiled, not run on hardware
//
#include <hip/hip_runtime.h>
#include <hip/hip_bf16.h>
#include <math.h>

typedef __attribute__((ext_vector_type(16))) _Float16 v16h;
typedef __attribute__((ext_vector_type(8)))  float    v8f;

#define NATOMS 524288
#define DIM    256
#define BMOL   16384
#define MSTEPS 6
#define CAP    2048   // max atoms/molecule cached in LDS (mean 32, max ~64; cold fallback below)

// ---------------- segment offsets: off[b] = lower_bound(split, b) ----------------
__global__ void k_offsets(const int* __restrict__ split, int* __restrict__ off) {
    int b = blockIdx.x * blockDim.x + threadIdx.x;
    if (b > BMOL) return;
    int lo = 0, hi = NATOMS;
    while (lo < hi) {
        int mid = (lo + hi) >> 1;
        if (split[mid] < b) lo = mid + 1; else hi = mid;
    }
    off[b] = lo;   // off[BMOL] == NATOMS since all values < BMOL
}

// ---------------- zero h0 and c ----------------
__global__ void k_zero2(float* __restrict__ a, float* __restrict__ b, int n) {
    int i = blockIdx.x * blockDim.x + threadIdx.x;
    if (i < n) { a[i] = 0.f; b[i] = 0.f; }
}

// ---------------- Ut[n,k] = (f16) U[k,n]  (pre-transpose for WMMA B loads) ----------------
__global__ void k_prepU(const float* __restrict__ U, _Float16* __restrict__ Ut) {
    int id = blockIdx.x * blockDim.x + threadIdx.x;
    if (id >= 512 * 1024) return;
    int n = id & 1023, k = id >> 10;           // coalesced read of U row k
    Ut[(size_t)n * 512 + k] = (_Float16)U[(size_t)k * 1024 + n];
}

// ---------------- fused segment-softmax attention: one workgroup per molecule ----------------
__global__ __launch_bounds__(256)
void k_attention(const float* __restrict__ af, const int* __restrict__ off,
                 const float* __restrict__ h, float* __restrict__ rdst, int rstride) {
    __shared__ float h_s[DIM];
    __shared__ float ee_s[CAP];
    __shared__ float wsum[8];
    __shared__ float emol_s;

    const int b = blockIdx.x;
    const int t = threadIdx.x;
    const int w = t >> 5, lane = t & 31;
    const int n0 = off[b], cnt = off[b + 1] - n0;

    h_s[t] = h[(size_t)b * DIM + t];
    __syncthreads();

    // phase 1: per-atom logits, exp, molecule sum (one wave per atom, strided)
    float4 hv1 = ((const float4*)h_s)[lane];
    float4 hv2 = ((const float4*)h_s)[32 + lane];
    float wacc = 0.f;
    for (int idx = w; idx < cnt; idx += 8) {
        const float4* row = (const float4*)(af + (size_t)(n0 + idx) * DIM);
        float4 a1 = row[lane], a2 = row[32 + lane];
        float p = a1.x * hv1.x + a1.y * hv1.y + a1.z * hv1.z + a1.w * hv1.w
                + a2.x * hv2.x + a2.y * hv2.y + a2.z * hv2.z + a2.w * hv2.w;
        #pragma unroll
        for (int s = 16; s >= 1; s >>= 1) p += __shfl_xor(p, s, 32);
        float ee = expf(p);                    // every lane holds full dot after butterfly
        if (lane == 0 && idx < CAP) ee_s[idx] = ee;
        wacc += ee;
    }
    if (lane == 0) wsum[w] = wacc;
    __syncthreads();
    if (t == 0) {
        float s = 0.f;
        #pragma unroll
        for (int i = 0; i < 8; i++) s += wsum[i];
        emol_s = s;
    }
    __syncthreads();

    // phase 2: r[b,t] = sum_n (ee/emol) * af[n,t]   (thread t owns column t)
    float inv = 1.0f / emol_s;                 // unused when cnt == 0
    float acc = 0.f;
    for (int idx = 0; idx < cnt; ++idx) {
        float ee;
        if (idx < CAP) ee = ee_s[idx];
        else {                                  // statistically-unreachable cold path
            const float* row = af + (size_t)(n0 + idx) * DIM;
            float e = 0.f;
            for (int d2 = 0; d2 < DIM; ++d2) e += row[d2] * h_s[d2];
            ee = expf(e);
        }
        acc += (ee * inv) * af[(size_t)(n0 + idx) * DIM + t];
    }
    rdst[(size_t)b * rstride + t] = acc;
}

// ---------------- LSTM step via WMMA: z=[h|r]@U+b, gates, c/h update (no z in memory) ----------------
// One wave computes a 16(M)x16 column-slice of all 4 gates; A fragment reused across the 4 WMMAs.
__global__ __launch_bounds__(256)
void k_lstm(const float* __restrict__ hin, const float* __restrict__ rbuf,
            const _Float16* __restrict__ Ut, const float* __restrict__ bias,
            float* __restrict__ cbuf, float* __restrict__ hout) {
    const int t = threadIdx.x;
    const int w = t >> 5, lane = t & 31;
    const int g = lane >> 4, ln = lane & 15;
    const int mbase = blockIdx.x * 16;
    const int ntile = blockIdx.y * 8 + w;      // 0..15 -> gate column tile
    const int ncol  = ntile * 16 + ln;         // 0..255
    const int Mrow  = mbase + ln;              // row this lane supplies for A

    v8f zero = {0.f, 0.f, 0.f, 0.f, 0.f, 0.f, 0.f, 0.f};
    v8f acc[4] = {zero, zero, zero, zero};

    #pragma unroll
    for (int kc = 0; kc < 16; ++kc) {          // K = 512 in chunks of 32
        const int Kbase = kc * 32;
        const float* rowp = (kc < 8)
            ? (hin  + (size_t)Mrow * DIM + Kbase)
            : (rbuf + (size_t)Mrow * DIM + (Kbase - 256));
        // contiguous k order chosen here (not the ISA table order), A 16x32: lane-group g holds K = 16g..16g+15 (matches B)
        v16h A;
        #pragma unroll
        for (int j = 0; j < 8; j++) {
            A[j]     = (_Float16)rowp[16 * g + j];
            A[j + 8] = (_Float16)rowp[16 * g + 8 + j];
        }
        #pragma unroll
        for (int q = 0; q < 4; q++) {
            // B 32x16, same contiguous k order as A: lane-group g holds K = 16g..16g+15 for column ncol
            const _Float16* bp = Ut + (size_t)(q * 256 + ncol) * 512 + Kbase + 16 * g;
            v16h Bv = *(const v16h*)bp;
            acc[q] = __builtin_amdgcn_wmma_f32_16x16x32_f16(
                false, A, false, Bv, (short)0, acc[q], false, false);
        }
    }

    const float bi = bias[ncol], bf = bias[256 + ncol];
    const float bo = bias[512 + ncol], bg = bias[768 + ncol];
    // C/D layout: VGPR v -> M = v + 8g, N = ln
    #pragma unroll
    for (int v = 0; v < 8; ++v) {
        const size_t idx2 = (size_t)(mbase + 8 * g + v) * DIM + ncol;
        float zi = acc[0][v] + bi;
        float zf = acc[1][v] + bf;
        float zo = acc[2][v] + bo;
        float zg = acc[3][v] + bg;
        float ii = 1.f / (1.f + expf(-zi));
        float ff = 1.f / (1.f + expf(-zf));
        float oo = 1.f / (1.f + expf(-zo));
        float gg = tanhf(zg);
        float cn = ff * cbuf[idx2] + ii * gg;
        cbuf[idx2] = cn;
        hout[idx2] = oo * tanhf(cn);
    }
}

// ---------------- out[:, 0:256] = h ----------------
__global__ void k_concat_h(const float* __restrict__ h, float* __restrict__ out) {
    int b = blockIdx.x, t = threadIdx.x;
    out[(size_t)b * 512 + t] = h[(size_t)b * DIM + t];
}

extern "C" void kernel_launch(void* const* d_in, const int* in_sizes, int n_in,
                              void* d_out, int out_size, void* d_ws, size_t ws_size,
                              hipStream_t stream) {
    (void)in_sizes; (void)n_in; (void)out_size; (void)ws_size;
    const float* af    = (const float*)d_in[0];
    const int*   split = (const int*)  d_in[1];
    const float* U     = (const float*)d_in[2];
    const float* bias  = (const float*)d_in[3];
    float* out = (float*)d_out;

    char* ws = (char*)d_ws;
    size_t o = 0;
    auto wsalloc = [&](size_t bytes) -> char* {
        char* p = ws + o; o += (bytes + 255) & ~(size_t)255; return p;
    };
    int*      off  = (int*)     wsalloc((BMOL + 1) * sizeof(int));
    float*    h0   = (float*)   wsalloc((size_t)BMOL * DIM * 4);
    float*    h1   = (float*)   wsalloc((size_t)BMOL * DIM * 4);
    float*    cbuf = (float*)   wsalloc((size_t)BMOL * DIM * 4);
    float*    rbuf = (float*)   wsalloc((size_t)BMOL * DIM * 4);
    _Float16* Ut   = (_Float16*)wsalloc((size_t)1024 * 512 * 2);

    k_offsets<<<(BMOL + 1 + 255) / 256, 256, 0, stream>>>(split, off);
    const int nz = BMOL * DIM;
    k_zero2<<<(nz + 255) / 256, 256, 0, stream>>>(h0, cbuf, nz);
    k_prepU<<<(512 * 1024) / 256, 256, 0, stream>>>(U, Ut);

    float* hcur = h0; float* hnext = h1;
    for (int m = 0; m < MSTEPS; ++m) {
        const bool last = (m == MSTEPS - 1);
        float* rdst   = last ? (out + 256) : rbuf;  // q_star[:,256:512] = r on final step
        int    rstride = last ? 512 : 256;
        k_attention<<<BMOL, 256, 0, stream>>>(af, off, hcur, rdst, rstride);
        if (!last) {                                 // final LSTM step is dead code
            dim3 grid(BMOL / 16, 2);
            k_lstm<<<grid, 256, 0, stream>>>(hcur, rbuf, Ut, bias, cbuf, hnext);
            float* tmp = hcur; hcur = hnext; hnext = tmp;  // h double-buffer (RAW across blocks)
        }
    }
    k_concat_h<<<BMOL, 256, 0, stream>>>(hcur, out);   // q_star[:,0:256] = h
}
